// HTSATModel_15247133900946
// MI455X (gfx1250) — matrix operand order repaired and checked by reading, not compiled, not run on hardware
//
#include <hip/hip_runtime.h>
#include <cstdint>
#include <cstddef>

typedef _Float16 half_t;
typedef __attribute__((ext_vector_type(16))) _Float16 v16h;
typedef __attribute__((ext_vector_type(8)))  _Float16 v8h;
typedef __attribute__((ext_vector_type(8)))  float    v8f;
typedef int v4i_t __attribute__((vector_size(16)));

#define DEV __device__ __forceinline__
#define AS_GLOBAL __attribute__((address_space(1)))
#define AS_SHARED __attribute__((address_space(3)))

// ---------------- static problem config (HTSAT stage-1 Swin block) ----------
// B=64, H=W=64, C=96, WS=8, SS=4, NH=4, hd=24, N=64 tok/window
// tokens = 262144, windows = 4096

// ---------------------------------------------------------------------------
// Async global->LDS copy (gfx1250 GLOBAL_LOAD_ASYNC_TO_LDS_B128, ASYNCcnt),
// with a plain b128 copy fallback so the compile cannot break.
// Builtin signature (from hipcc diagnostic): param0 = AS1 int4* (non-const).
// ---------------------------------------------------------------------------
#if defined(__has_builtin)
#if __has_builtin(__builtin_amdgcn_global_load_async_to_lds_b128) && \
    __has_builtin(__builtin_amdgcn_s_wait_asynccnt)
#define HTSAT_ASYNC_LDS 1
#endif
#endif

DEV void copy16_g2l(half_t* lds, const half_t* g) {
#ifdef HTSAT_ASYNC_LDS
    void* gnc = const_cast<half_t*>(g);
    __builtin_amdgcn_global_load_async_to_lds_b128(
        (AS_GLOBAL v4i_t*)gnc, (AS_SHARED v4i_t*)lds, 0, 0);
#else
    *(float4*)lds = *(const float4*)g;
#endif
}
DEV void wait_g2l() {
#ifdef HTSAT_ASYNC_LDS
    __builtin_amdgcn_s_wait_asynccnt(0);
#endif
}

// ---------------------------------------------------------------------------
// f32 -> f16 elementwise convert (weight prep; W native [N][K] layout is the
// B-matrix layout the GEMM wants, so no transpose needed)
// ---------------------------------------------------------------------------
__global__ void htsat_cvt_f16(const float* __restrict__ src, half_t* __restrict__ dst, int n) {
    int i = blockIdx.x * 256 + threadIdx.x;
    if (i < n) dst[i] = (half_t)src[i];
}

// ---------------------------------------------------------------------------
// Precompute bias+mask table: bmat[widx][h][i][j] =
//   rpb[relpos(i,j)][h] + (same shift-class ? 0 : -100)
// 64*4*64*64 f32 = 4MB -> lives in the 192MB L2 for the attention pass.
// ---------------------------------------------------------------------------
DEV int swin_cls(int p) { return p < 56 ? 0 : (p < 60 ? 1 : 2); }

__global__ __launch_bounds__(256)
void htsat_build_bias(const float* __restrict__ rpb, float* __restrict__ bmat) {
    int idx = blockIdx.x * 256 + threadIdx.x;       // < 1048576
    int j = idx & 63, i = (idx >> 6) & 63, h = (idx >> 12) & 3, widx = idx >> 14;
    int yi = i >> 3, xi = i & 7, yj = j >> 3, xj = j & 7;
    int ridx = (yi - yj + 7) * 15 + (xi - xj + 7);
    int wy = widx >> 3, wx = widx & 7;
    int ci = swin_cls(wy * 8 + yi) * 3 + swin_cls(wx * 8 + xi);
    int cj = swin_cls(wy * 8 + yj) * 3 + swin_cls(wx * 8 + xj);
    bmat[idx] = rpb[ridx * 4 + h] + ((ci == cj) ? 0.0f : -100.0f);
}

// ---------------------------------------------------------------------------
// LayerNorm over C=96, wave-per-token (wave32: 3 elems/lane), optionally fused
// with cyclic shift (-4,-4) + window partition permutation.
// ---------------------------------------------------------------------------
__global__ __launch_bounds__(256)
void htsat_layernorm(const float* __restrict__ xin, const float* __restrict__ g,
                     const float* __restrict__ b, half_t* __restrict__ out, int windowed) {
    const int wv = threadIdx.x >> 5, lane = threadIdx.x & 31;
    const int m = blockIdx.x * 8 + wv;             // dest token (window-ordered if windowed)
    size_t src;
    if (windowed) {
        int bi = m >> 12, rem = m & 4095;
        int win = rem >> 6, tok = rem & 63;
        int row = (win >> 3) * 8 + (tok >> 3);     // shifted-image coords
        int col = (win & 7) * 8 + (tok & 7);
        int r = (row + 4) & 63, c = (col + 4) & 63; // roll(x,-4): shifted[i]=x[(i+4)%64]
        src = (size_t)bi * 4096 + (size_t)r * 64 + c;
    } else {
        src = (size_t)m;
    }
    const float* xp = xin + src * 96;
    float v0 = xp[lane], v1 = xp[lane + 32], v2 = xp[lane + 64];
    float s = v0 + v1 + v2;
#pragma unroll
    for (int msk = 16; msk >= 1; msk >>= 1) s += __shfl_xor(s, msk, 32);
    float mu = s * (1.0f / 96.0f);
    float d0 = v0 - mu, d1 = v1 - mu, d2 = v2 - mu;
    float q = d0 * d0 + d1 * d1 + d2 * d2;
#pragma unroll
    for (int msk = 16; msk >= 1; msk >>= 1) q += __shfl_xor(q, msk, 32);
    float rs = rsqrtf(q * (1.0f / 96.0f) + 1e-5f);
    half_t* op = out + (size_t)m * 96;
    op[lane]      = (half_t)(d0 * rs * g[lane]      + b[lane]);
    op[lane + 32] = (half_t)(d1 * rs * g[lane + 32] + b[lane + 32]);
    op[lane + 64] = (half_t)(d2 * rs * g[lane + 64] + b[lane + 64]);
}

// ---------------------------------------------------------------------------
// WMMA fragment helpers (gfx1250 wave32 layouts, CDNA5 ISA)
// A 16x32 f16: lane L holds row M=L%16; halves 0..7 = K=(L<16?0:8)+0..7,
//              halves 8..15 = K=16+(L<16?0:8)+0..7      -> two ds_load_b128
// B 32x16 f16: lane L holds col N=L%16; same K order as A (tile stored
//              [N][K] in LDS)                            -> two ds_load_b128
// C/D 16x16 f32: lane L, reg r -> (M = r + (L<16?0:8), N = L%16)
// ---------------------------------------------------------------------------
DEV v16h frag_cat(v8h lo, v8h hi) {
    return __builtin_shufflevector(lo, hi, 0,1,2,3,4,5,6,7,8,9,10,11,12,13,14,15);
}
DEV v8f wmma_f16(v16h a, v16h b, v8f c) {
    return __builtin_amdgcn_wmma_f32_16x16x32_f16(false, a, false, b, (short)0, c, false, false);
}

// ---------------------------------------------------------------------------
// Generic WMMA GEMM: out(M x N) = A(M x K, f16 row-major) * W(N x K, f16)^T
// Block tile 128x96, 8 waves (wave = 16 rows x 96 cols = 6 accumulators).
// K staged by 96 through LDS (one barrier pair per 3 WMMA K-steps); staging
// uses async global->LDS b128 when available. Epilogue templated (adds bias).
// ---------------------------------------------------------------------------
template <typename Epi>
__global__ __launch_bounds__(256)
void htsat_wmma_gemm(const half_t* __restrict__ A, const half_t* __restrict__ W,
                     const float* __restrict__ bias, int K, Epi epi) {
    __shared__ half_t As[128][96];   // 24KB
    __shared__ half_t Bs[96][96];    // 18KB
    const int m0 = blockIdx.x * 128;
    const int n0 = blockIdx.y * 96;
    const int tid = threadIdx.x;
    const int wv = tid >> 5, lane = tid & 31;
    const int lm = lane & 15, hi = lane >> 4;

    v8f acc[6];
#pragma unroll
    for (int c = 0; c < 6; ++c) acc[c] = (v8f){0.f,0.f,0.f,0.f,0.f,0.f,0.f,0.f};

    for (int k0 = 0; k0 < K; k0 += 96) {
        // stage A tile 128x96: 1536 x 16B chunks, 6 per thread
        for (int s = tid; s < 1536; s += 256) {
            int r = s / 12, cc = s % 12;
            copy16_g2l(&As[r][cc * 8], A + (size_t)(m0 + r) * K + k0 + cc * 8);
        }
        // stage B tile 96x96 ([N][K] native): 1152 x 16B chunks
        for (int s = tid; s < 1152; s += 256) {
            int r = s / 12, cc = s % 12;
            copy16_g2l(&Bs[r][cc * 8], W + (size_t)(n0 + r) * K + k0 + cc * 8);
        }
        wait_g2l();
        __syncthreads();

        const int row = wv * 16 + lm, klo = hi * 8, kb = hi * 16;
#pragma unroll
        for (int kk = 0; kk < 3; ++kk) {
            v16h af = frag_cat(*(const v8h*)&As[row][kk * 32 + klo],
                               *(const v8h*)&As[row][kk * 32 + 16 + klo]);
#pragma unroll
            for (int c = 0; c < 6; ++c) {
                const int n = c * 16 + lm;
                v16h bf = frag_cat(*(const v8h*)&Bs[n][kk * 32 + klo],
                                   *(const v8h*)&Bs[n][kk * 32 + 16 + klo]);
                acc[c] = wmma_f16(af, bf, acc[c]);
            }
        }
        __syncthreads();
    }
#pragma unroll
    for (int c = 0; c < 6; ++c)
#pragma unroll
        for (int r = 0; r < 8; ++r) {
            int m = m0 + wv * 16 + hi * 8 + r;
            int n = n0 + c * 16 + lm;
            epi(m, n, acc[c][r] + bias[n]);
        }
}

// ------------------------------ epilogues ----------------------------------
struct EpiH16 {                      // plain f16 store (QKV)
    half_t* out; int ldo;
    DEV void operator()(int m, int n, float v) const {
        out[(size_t)m * ldo + n] = (half_t)v;
    }
};
struct EpiGeluH16 {                  // exact GELU -> f16 (fc1)
    half_t* out; int ldo;
    DEV void operator()(int m, int n, float v) const {
        float gv = 0.5f * v * (1.0f + erff(v * 0.70710678118654752f));
        out[(size_t)m * ldo + n] = (half_t)gv;
    }
};
struct EpiProjRes {                  // proj: window-reverse + roll(+4) + residual
    const float* x; float* x2;
    DEV void operator()(int m, int n, float v) const {
        int bi = m >> 12, rem = m & 4095;
        int win = rem >> 6, tok = rem & 63;
        int row = (win >> 3) * 8 + (tok >> 3);
        int col = (win & 7) * 8 + (tok & 7);
        int r = (row + 4) & 63, c = (col + 4) & 63;
        size_t t = ((size_t)bi * 4096 + (size_t)r * 64 + c) * 96 + n;
        x2[t] = x[t] + v;
    }
};
struct EpiFc2Out {                   // fc2: + residual x2 -> final f32 output
    const float* x2; float* out;
    DEV void operator()(int m, int n, float v) const {
        size_t i = (size_t)m * 96 + n;
        out[i] = x2[i] + v;
    }
};

// ---------------------------------------------------------------------------
// Windowed attention, one block per (window, head). 4 waves, 128 threads.
// q*scale, k staged [64][32] (d padded 24->32), v staged transposed [32][64].
// scores via 16 WMMAs; bias+mask from precomputed L2-resident table;
// softmax with 16-lane shfl reductions; attn through LDS; attn@v via 16 WMMAs.
// ---------------------------------------------------------------------------
__global__ __launch_bounds__(128)
void htsat_attn(const half_t* __restrict__ qkv, const float* __restrict__ bmat,
                half_t* __restrict__ obuf) {
    __shared__ half_t qs[64][32];
    __shared__ half_t ks[64][32];
    __shared__ half_t vt[32][64];
    __shared__ half_t at[64][64];
    const int w = blockIdx.x, h = blockIdx.y;
    const int tid = threadIdx.x;
    const float scale = 0.20412414523193150818f;   // 24^-0.5
    const size_t base = (size_t)w * 64 * 288;

    for (int i = tid; i < 2048; i += 128) {
        int tok = i >> 5, d = i & 31;
        const size_t o = base + (size_t)tok * 288 + h * 24 + d;
        float qv = (d < 24) ? (float)qkv[o] * scale : 0.0f;
        qs[tok][d] = (half_t)qv;
        ks[tok][d] = (d < 24) ? qkv[o + 96]  : (half_t)0.0f;
        vt[d][tok] = (d < 24) ? qkv[o + 192] : (half_t)0.0f;
    }
    __syncthreads();

    const int wv = tid >> 5, lane = tid & 31, lm = lane & 15, hi = lane >> 4;

    // ---- scores = (q*scale) @ k^T : 4 N-tiles, K=32 (padded) ----
    v8f sc[4];
    {
        const int row = wv * 16 + lm, klo = hi * 8, kb = hi * 16;
        v16h af = frag_cat(*(const v8h*)&qs[row][klo], *(const v8h*)&qs[row][16 + klo]);
#pragma unroll
        for (int nt = 0; nt < 4; ++nt) {
            const int j = nt * 16 + lm;
            v16h bf = frag_cat(*(const v8h*)&ks[j][klo], *(const v8h*)&ks[j][16 + klo]);
            v8f z = (v8f){0.f,0.f,0.f,0.f,0.f,0.f,0.f,0.f};
            sc[nt] = wmma_f16(af, bf, z);
        }
    }

    // ---- + (rel-pos bias + shift mask) table, softmax over 64 cols ----
    const float* bm = bmat + ((((size_t)(w & 63)) * 4 + h) << 12);   // [i][j] 64x64
#pragma unroll
    for (int r = 0; r < 8; ++r) {
        const int i = wv * 16 + hi * 8 + r;
        float vals[4];
        float rmax = -1e30f;
#pragma unroll
        for (int nt = 0; nt < 4; ++nt) {
            float v = sc[nt][r] + bm[i * 64 + nt * 16 + lm];
            vals[nt] = v;
            rmax = fmaxf(rmax, v);
        }
#pragma unroll
        for (int msk = 8; msk >= 1; msk >>= 1) rmax = fmaxf(rmax, __shfl_xor(rmax, msk, 32));
        float sum = 0.0f;
#pragma unroll
        for (int nt = 0; nt < 4; ++nt) { vals[nt] = __expf(vals[nt] - rmax); sum += vals[nt]; }
#pragma unroll
        for (int msk = 8; msk >= 1; msk >>= 1) sum += __shfl_xor(sum, msk, 32);
        const float inv = 1.0f / sum;
#pragma unroll
        for (int nt = 0; nt < 4; ++nt) at[i][nt * 16 + lm] = (half_t)(vals[nt] * inv);
    }
    __syncthreads();

    // ---- out = attn @ v : K=64 (2 steps), 2 N-tiles (d padded 24->32) ----
    v8f ov[2];
    ov[0] = (v8f){0.f,0.f,0.f,0.f,0.f,0.f,0.f,0.f};
    ov[1] = ov[0];
#pragma unroll
    for (int ks2 = 0; ks2 < 2; ++ks2) {
        const int row = wv * 16 + lm, klo = ks2 * 32 + hi * 8, kb = ks2 * 32 + hi * 16;
        v16h af = frag_cat(*(const v8h*)&at[row][klo], *(const v8h*)&at[row][16 + klo]);
#pragma unroll
        for (int nt = 0; nt < 2; ++nt) {
            const int d = nt * 16 + lm;
            v16h bf = frag_cat(*(const v8h*)&vt[d][klo], *(const v8h*)&vt[d][16 + klo]);
            ov[nt] = wmma_f16(af, bf, ov[nt]);
        }
    }
#pragma unroll
    for (int nt = 0; nt < 2; ++nt)
#pragma unroll
        for (int r = 0; r < 8; ++r) {
            const int d = nt * 16 + lm;
            if (d < 24) {
                const int tok = wv * 16 + hi * 8 + r;
                obuf[((size_t)w * 64 + tok) * 96 + h * 24 + d] = (half_t)ov[nt][r];
            }
        }
}

// ---------------------------------------------------------------------------
// host-side orchestration
// ---------------------------------------------------------------------------
extern "C" void kernel_launch(void* const* d_in, const int* in_sizes, int n_in,
                              void* d_out, int out_size, void* d_ws, size_t ws_size,
                              hipStream_t stream) {
    (void)in_sizes; (void)n_in; (void)out_size; (void)ws_size;
    const float* x      = (const float*)d_in[0];
    const float* n1g    = (const float*)d_in[1];
    const float* n1b    = (const float*)d_in[2];
    const float* qkv_w  = (const float*)d_in[3];
    const float* qkv_b  = (const float*)d_in[4];
    const float* rpb    = (const float*)d_in[5];
    const float* proj_w = (const float*)d_in[6];
    const float* proj_b = (const float*)d_in[7];
    const float* n2g    = (const float*)d_in[8];
    const float* n2b    = (const float*)d_in[9];
    const float* fc1_w  = (const float*)d_in[10];
    const float* fc1_b  = (const float*)d_in[11];
    const float* fc2_w  = (const float*)d_in[12];
    const float* fc2_b  = (const float*)d_in[13];
    float* out = (float*)d_out;

    char* ws = (char*)d_ws;
    half_t* wq   = (half_t*)(ws + 0);           // 288x96 f16
    half_t* wp   = (half_t*)(ws + 55296);       // 96x96
    half_t* w1   = (half_t*)(ws + 73728);       // 384x96
    half_t* w2   = (half_t*)(ws + 147456);      // 96x384
    float*  bmat = (float*) (ws + 221184);      // 64x4x64x64 f32 (4MB)
    half_t* hwin = (half_t*)(ws + 4415488);     // 262144x96 f16 (later reused: obuf)
    half_t* qkvb = (half_t*)(ws + 54747136);    // 262144x288 f16 (later reused: h2)
    float*  x2   = (float*) (ws + 205742080);   // 262144x96 f32
    half_t* f1   = (half_t*)(ws + 306405376);   // 262144x384 f16
    half_t* obuf = hwin;                        // dead after QKV GEMM
    half_t* h2   = qkvb;                        // dead after attention

    htsat_cvt_f16<<<(27648 + 255) / 256, 256, 0, stream>>>(qkv_w, wq, 27648);
    htsat_cvt_f16<<<(9216  + 255) / 256, 256, 0, stream>>>(proj_w, wp, 9216);
    htsat_cvt_f16<<<(36864 + 255) / 256, 256, 0, stream>>>(fc1_w, w1, 36864);
    htsat_cvt_f16<<<(36864 + 255) / 256, 256, 0, stream>>>(fc2_w, w2, 36864);
    htsat_build_bias<<<4096, 256, 0, stream>>>(rpb, bmat);

    // LN1 + shift + window partition
    htsat_layernorm<<<32768, 256, 0, stream>>>(x, n1g, n1b, hwin, 1);
    // QKV: [262144x96] @ [96x288]
    htsat_wmma_gemm<<<dim3(2048, 3), 256, 0, stream>>>(hwin, wq, qkv_b, 96, EpiH16{qkvb, 288});
    // windowed attention per (window, head)
    htsat_attn<<<dim3(4096, 4), 128, 0, stream>>>(qkvb, bmat, obuf);
    // proj + window reverse + roll + residual -> x2
    htsat_wmma_gemm<<<dim3(2048, 1), 256, 0, stream>>>(obuf, wp, proj_b, 96, EpiProjRes{x, x2});
    // LN2
    htsat_layernorm<<<32768, 256, 0, stream>>>(x2, n2g, n2b, h2, 0);
    // fc1 + GELU
    htsat_wmma_gemm<<<dim3(2048, 4), 256, 0, stream>>>(h2, w1, fc1_b, 96, EpiGeluH16{f1, 384});
    // fc2 + residual -> d_out
    htsat_wmma_gemm<<<dim3(2048, 1), 256, 0, stream>>>(f1, w2, fc2_b, 384, EpiFc2Out{x2, out});
}
